// ImprovedVectorQuantizer_16423954940664
// MI455X (gfx1250) — matrix operand order repaired and checked by reading, not compiled, not run on hardware
//
#include <hip/hip_runtime.h>
#include <hip/hip_bf16.h>

// ---------------------------------------------------------------------------
// VQ eval: distances = ||x||^2 + ||e||^2 - 2 x e^T ; argmin ; gather ; loss
// N=131072 tokens, D=64, K=1024 codes, fp32 in/out.
// Error-compensated bf16 split GEMM on v_wmma_f32_16x16x32_bf16 with 4-way
// M-register-blocking (64 rows/wave) so B-fragment LDS loads amortize over
// 24 WMMAs instead of 6; fused argmin + gather + loss.
// ---------------------------------------------------------------------------

#define N_TOKENS 131072
#define EMBED_DIM 64
#define NUM_CODES 1024
#define MTILES 4                              // row-tiles per wave
#define ROWS_PER_WAVE (16 * MTILES)           // 64
#define WAVES_PER_BLOCK 8
#define ROWS_PER_BLOCK (ROWS_PER_WAVE * WAVES_PER_BLOCK) // 512
#define SW 72  // padded LDS row stride in bf16 elements (144B: 16B-aligned, conflict-free)

#define XQ_ELEMS (N_TOKENS * EMBED_DIM)      // 8388608
#define LOSS_OFF XQ_ELEMS                    // 1 element
#define IDX_OFF (XQ_ELEMS + 1)               // 131072 elements

// LDS layout (bytes)
#define LDS_WHI 0
#define LDS_WLO (NUM_CODES * SW * 2)                      // 147456
#define LDS_ENORM (2 * NUM_CODES * SW * 2)                // 294912
#define LDS_IDX (LDS_ENORM + NUM_CODES * 4)               // 299008
#define LDS_BYTES (LDS_IDX + ROWS_PER_BLOCK * 4)          // 301056

typedef __attribute__((ext_vector_type(16))) __bf16 v16bf;
typedef __attribute__((ext_vector_type(8)))  float  v8f;

union Frag16 {
    v16bf v;
    uint4 q[2];
};

__global__ void vq_zero_loss_kernel(float* __restrict__ out) {
    if (threadIdx.x == 0 && blockIdx.x == 0) out[LOSS_OFF] = 0.0f;
}

__global__ __launch_bounds__(256, 1) void vq_main_kernel(const float* __restrict__ x,
                                                         const float* __restrict__ w,
                                                         float* __restrict__ out) {
    extern __shared__ char smem[];
    __bf16* __restrict__ whi = (__bf16*)(smem + LDS_WHI);
    __bf16* __restrict__ wlo = (__bf16*)(smem + LDS_WLO);
    float*  __restrict__ enormLds = (float*)(smem + LDS_ENORM);
    int*    __restrict__ idxLds   = (int*)(smem + LDS_IDX);

    const int tid = threadIdx.x;

    // ---- Phase 1: codebook -> LDS as bf16 hi/lo planes + e-norms (f32) ----
    #pragma unroll
    for (int rr = 0; rr < 4; ++rr) {
        const int n = tid * 4 + rr;
        const float* wr = w + n * EMBED_DIM;
        float s = 0.0f;
        #pragma unroll
        for (int k = 0; k < EMBED_DIM; k += 4) {
            float4 f = *(const float4*)(wr + k);
            float fv[4] = {f.x, f.y, f.z, f.w};
            #pragma unroll
            for (int i = 0; i < 4; ++i) {
                __bf16 hi = (__bf16)fv[i];
                whi[n * SW + k + i] = hi;
                wlo[n * SW + k + i] = (__bf16)(fv[i] - (float)hi);
                s += fv[i] * fv[i];
            }
        }
        enormLds[n] = s;
    }
    __syncthreads();

    // ---- Phase 2: per-wave 4x16-row tiles, sweep K=1024 codes -------------
    const int lane = tid & 31;
    const int wave = tid >> 5;
    const int r = lane & 15;   // column lane within half-wave
    const int h = lane >> 4;   // half-wave id
    const int rowBase = blockIdx.x * ROWS_PER_BLOCK + wave * ROWS_PER_WAVE;

    // A fragments (xh/xl, K-chunks 0..31 / 32..63) per documented 16-bit A
    // 16x32 layout: VGPR v<4 -> K=2v+8h, v>=4 -> K=16+2(v-4)+8h.
    Frag16 xh[MTILES][2], xl[MTILES][2];
    #pragma unroll
    for (int rt = 0; rt < MTILES; ++rt) {
        const float* xr = x + (rowBase + rt * 16 + r) * EMBED_DIM;
        #pragma unroll
        for (int c = 0; c < 2; ++c) {
            #pragma unroll
            for (int p = 0; p < 2; ++p) {
                const int ks = c * 32 + p * 16 + h * 8;
                float4 f0 = *(const float4*)(xr + ks);
                float4 f1 = *(const float4*)(xr + ks + 4);
                float fv[8] = {f0.x, f0.y, f0.z, f0.w, f1.x, f1.y, f1.z, f1.w};
                #pragma unroll
                for (int i = 0; i < 8; ++i) {
                    __bf16 hi = (__bf16)fv[i];
                    xh[rt][c].v[p * 8 + i] = hi;
                    xl[rt][c].v[p * 8 + i] = (__bf16)(fv[i] - (float)hi);
                }
            }
        }
    }

    float best[MTILES][8];
    int bidx[MTILES][8];
    #pragma unroll
    for (int rt = 0; rt < MTILES; ++rt)
        #pragma unroll
        for (int v = 0; v < 8; ++v) { best[rt][v] = 3.4e38f; bidx[rt][v] = 0; }

    // B fragment layout (16-bit 32x16), same K order as A: elements 0-7 hold
    // K=8h..8h+7, elements 8-15 hold K=16+8h..16+8h+7 of column n=r.
    for (int t = 0; t < NUM_CODES / 16; ++t) {
        const int n = t * 16 + r;
        const __bf16* bh = whi + n * SW + h * 8;
        const __bf16* bl = wlo + n * SW + h * 8;

        Frag16 wh0, wh1, wl0, wl1;
        wh0.q[0] = *(const uint4*)(bh);
        wh0.q[1] = *(const uint4*)(bh + 16);
        wh1.q[0] = *(const uint4*)(bh + 32);
        wh1.q[1] = *(const uint4*)(bh + 48);
        wl0.q[0] = *(const uint4*)(bl);
        wl0.q[1] = *(const uint4*)(bl + 16);
        wl1.q[0] = *(const uint4*)(bl + 32);
        wl1.q[1] = *(const uint4*)(bl + 48);

        const float en = enormLds[n];

        #pragma unroll
        for (int rt = 0; rt < MTILES; ++rt) {
            v8f acc = {0.f, 0.f, 0.f, 0.f, 0.f, 0.f, 0.f, 0.f};
            // dot = xh*wh + xh*wl + xl*wh (error-compensated split, f32 acc)
            acc = __builtin_amdgcn_wmma_f32_16x16x32_bf16(false, xh[rt][0].v, false, wh0.v, (short)0, acc, false, false);
            acc = __builtin_amdgcn_wmma_f32_16x16x32_bf16(false, xh[rt][1].v, false, wh1.v, (short)0, acc, false, false);
            acc = __builtin_amdgcn_wmma_f32_16x16x32_bf16(false, xl[rt][0].v, false, wh0.v, (short)0, acc, false, false);
            acc = __builtin_amdgcn_wmma_f32_16x16x32_bf16(false, xl[rt][1].v, false, wh1.v, (short)0, acc, false, false);
            acc = __builtin_amdgcn_wmma_f32_16x16x32_bf16(false, xh[rt][0].v, false, wl0.v, (short)0, acc, false, false);
            acc = __builtin_amdgcn_wmma_f32_16x16x32_bf16(false, xh[rt][1].v, false, wl1.v, (short)0, acc, false, false);

            // score = ||e||^2 - 2*dot (||x||^2 row-constant -> argmin-invariant)
            #pragma unroll
            for (int v = 0; v < 8; ++v) {
                const float s = __builtin_fmaf(-2.0f, acc[v], en);
                if (s < best[rt][v]) { best[rt][v] = s; bidx[rt][v] = n; }
            }
        }
    }

    // Cross-lane argmin over the 16 column-lanes of each half-wave.
    // Row M = v + 8*h of tile rt lives in VGPR v of lanes h*16..h*16+15.
    #pragma unroll
    for (int rt = 0; rt < MTILES; ++rt) {
        #pragma unroll
        for (int v = 0; v < 8; ++v) {
            #pragma unroll
            for (int d = 1; d < 16; d <<= 1) {
                float ob = __shfl_xor(best[rt][v], d, 32);
                int   oi = __shfl_xor(bidx[rt][v], d, 32);
                if (ob < best[rt][v] || (ob == best[rt][v] && oi < bidx[rt][v])) {
                    best[rt][v] = ob; bidx[rt][v] = oi;
                }
            }
            if (r == 0) idxLds[wave * ROWS_PER_WAVE + rt * 16 + h * 8 + v] = bidx[rt][v];
        }
    }
    __syncthreads();

    // ---- Phase 3: gather exact fp32 code rows, write x_quantized, loss ----
    float lsum = 0.0f;
    const int j = lane * 2;  // 32 lanes x 2 cols = 64
    #pragma unroll 4
    for (int m = 0; m < ROWS_PER_WAVE; ++m) {
        const int bi = idxLds[wave * ROWS_PER_WAVE + m];
        const int orow = rowBase + m;
        float2 wq = *(const float2*)(w + bi * EMBED_DIM + j);
        float2 xv = *(const float2*)(x + orow * EMBED_DIM + j);
        *(float2*)(out + orow * EMBED_DIM + j) = wq;
        const float d0 = wq.x - xv.x;
        const float d1 = wq.y - xv.y;
        lsum = __builtin_fmaf(d0, d0, lsum);
        lsum = __builtin_fmaf(d1, d1, lsum);
    }

    // Index output (reference returns int argmin; d_out is fp32 -> exact ints)
    #pragma unroll
    for (int q = 0; q < ROWS_PER_WAVE / 32; ++q) {
        const int m = q * 32 + lane;
        out[IDX_OFF + rowBase + m] = (float)idxLds[wave * ROWS_PER_WAVE + m];
    }

    // loss = (1 + COMMITMENT_COST) * mean((xq - x)^2)
    #pragma unroll
    for (int d = 1; d < 32; d <<= 1) lsum += __shfl_xor(lsum, d, 32);
    if (lane == 0) {
        atomicAdd(out + LOSS_OFF, lsum * (1.25f / (float)XQ_ELEMS));
    }
}

extern "C" void kernel_launch(void* const* d_in, const int* in_sizes, int n_in,
                              void* d_out, int out_size, void* d_ws, size_t ws_size,
                              hipStream_t stream) {
    (void)in_sizes; (void)n_in; (void)d_ws; (void)ws_size; (void)out_size;
    const float* x = (const float*)d_in[0];
    const float* w = (const float*)d_in[1];
    float* out = (float*)d_out;

    // ~294 KB of the WGP's 320 KB LDS -> raise the dynamic-LDS cap.
    hipFuncSetAttribute((const void*)vq_main_kernel,
                        hipFuncAttributeMaxDynamicSharedMemorySize, LDS_BYTES);

    vq_zero_loss_kernel<<<1, 1, 0, stream>>>(out);
    vq_main_kernel<<<N_TOKENS / ROWS_PER_BLOCK, 256, LDS_BYTES, stream>>>(x, w, out);
}
